// HybridClassicalFilterSNN_65352222376607
// MI455X (gfx1250) — matrix operand order repaired and checked by reading, not compiled, not run on hardware
//
#include <hip/hip_runtime.h>

typedef __attribute__((ext_vector_type(16))) _Float16 v16h;
typedef __attribute__((ext_vector_type(8)))  _Float16 h8;
typedef __attribute__((ext_vector_type(8)))  float    v8f;
typedef __attribute__((ext_vector_type(4)))  unsigned int u32x4;
typedef __attribute__((ext_vector_type(8)))  int          i32x8;
typedef __attribute__((ext_vector_type(4)))  int          i32x4;

#define BZ 32
#define HZ 64
#define WZ 64
#define TZ 16
#define C1 64
#define C2 128
#define NPIX (BZ*HZ*WZ)      /* 131072 */
#define ROWH (WZ*C1)         /* 4096 halves = 8192 B per (b,h) row of s1 */

#define GLOBAL_AS __attribute__((address_space(1)))
#define SHARED_AS __attribute__((address_space(3)))

#if __has_builtin(__builtin_amdgcn_tensor_load_to_lds) && __has_builtin(__builtin_amdgcn_s_wait_tensorcnt)
#define HAVE_TDM 1
#endif
#if __has_builtin(__builtin_amdgcn_global_load_async_to_lds_b128) && __has_builtin(__builtin_amdgcn_s_wait_asynccnt)
#define HAVE_ASYNC 1
#endif

// ---------------------------------------------------------------------------
// Weight reformat: conv_w [3,3,64,128] f32 (HWIO) -> pre-swizzled f16 WMMA
// B-fragments (lane L reads 16 contiguous f16 at frag + L*16).
// ---------------------------------------------------------------------------
__global__ void reformat_w_kernel(const float* __restrict__ conv_w,
                                  _Float16* __restrict__ wB) {
    int d = blockIdx.x * 256 + threadIdx.x;
    if (d >= 9 * 2 * 8 * 512) return;
    int j    =  d        & 15;
    int lane = (d >> 4)  & 31;
    int nt   = (d >> 9)  & 7;
    int kc   = (d >> 12) & 1;
    int tap  =  d >> 13;
    int kin  = kc * 32 + (j & 7) + ((j >> 3) << 4) + ((lane >> 4) << 3);
    int n    = nt * 16 + (lane & 15);
    wB[d] = (_Float16)conv_w[(tap * 64 + kin) * 128 + n];
}

// ---------------------------------------------------------------------------
// k1: classical filter bank + 1x1 fuse conv + LIF1. One thread per pixel.
// ---------------------------------------------------------------------------
__global__ void k1_fuse_lif1(const float* __restrict__ x_seq,
                             const float* __restrict__ fuse_w,
                             const float* __restrict__ fuse_b,
                             float* __restrict__ u1,
                             _Float16* __restrict__ s1,
                             float* __restrict__ accs, int t) {
    int p = blockIdx.x * 256 + threadIdx.x;
    int b = p >> 12;
    int h = (p >> 6) & 63;
    int w = p & 63;
    const float* xt = x_seq + (size_t)t * NPIX * 2;

    float g[3][3];
    #pragma unroll
    for (int dy = -1; dy <= 1; ++dy) {
        #pragma unroll
        for (int dx = -1; dx <= 1; ++dx) {
            int hh = h + dy, ww = w + dx;
            float v = 0.0f;
            if (hh >= 0 && hh < HZ && ww >= 0 && ww < WZ) {
                const float* q = xt + (((size_t)b * HZ + hh) * WZ + ww) * 2;
                v = 0.5f * (q[0] + q[1]);
            }
            g[dy + 1][dx + 1] = v;
        }
    }
    float r0 = -g[0][0] + g[0][2] - 2.f*g[1][0] + 2.f*g[1][2] - g[2][0] + g[2][2];
    float r1 = -g[0][0] - 2.f*g[0][1] - g[0][2] + g[2][0] + 2.f*g[2][1] + g[2][2];
    float r2 =  g[0][1] + g[1][0] - 4.f*g[1][1] + g[1][2] + g[2][1];
    float en = r0*r0 + r1*r1 + r2*r2;

    const float* qc = xt + (((size_t)b * HZ + h) * WZ + w) * 2;
    float xf[5] = { qc[0], qc[1], r0, r1, r2 };

    float ssum = 0.0f;
    size_t base = (size_t)p * C1;
    for (int c = 0; c < C1; ++c) {
        float i1 = fuse_b[c];
        #pragma unroll
        for (int k = 0; k < 5; ++k) i1 += xf[k] * fuse_w[k * C1 + c];
        float u = 0.9f * u1[base + c] + i1;
        float s = (u - 1.0f > 0.0f) ? 1.0f : 0.0f;
        u1[base + c] = u - s;
        s1[base + c] = (_Float16)s;
        ssum += s;
    }

    __shared__ float sE, sS;
    if (threadIdx.x == 0) { sE = 0.f; sS = 0.f; }
    __syncthreads();
    atomicAdd(&sE, en);
    atomicAdd(&sS, ssum);
    __syncthreads();
    if (threadIdx.x == 0) {
        atomicAdd(&accs[t], sE);
        atomicAdd(&accs[TZ + t], sS);
    }
}

// ---------------------------------------------------------------------------
// k2: 3x3 conv 64->128 as implicit-GEMM WMMA f16 (K=576), A staged in LDS via
// the CDNA5 TDM / async-to-LDS path, fused LIF2 + spike reduction.
// Block = 256 thr = 8 waves; tile = 64 pixels (one (b,h) row) x 128 channels.
// LDS tile: 3 rows (h-1,h,h+1) x 64 pixels x 64 ch f16 = 24 KB.
// ---------------------------------------------------------------------------
__global__ void k2_conv_lif2(const _Float16* __restrict__ s1,
                             const _Float16* __restrict__ wB,
                             const float* __restrict__ conv_b,
                             float* __restrict__ u2,
                             float* __restrict__ sumS2,
                             float* __restrict__ accs, int t) {
    __shared__ _Float16 tileA[3 * ROWH];   // 24576 B
    __shared__ float cSum[C2];

    const int tid    = threadIdx.x;
    const int lane   = tid & 31;
    const int wave   = tid >> 5;
    const int m_tile = wave & 3;
    const int n_half = wave >> 2;

    const int pix0 = blockIdx.x * 64;
    const int b    = blockIdx.x >> 6;
    const int h    = blockIdx.x & 63;
    const int mrow = lane & 15;
    const int w    = m_tile * 16 + mrow;

    // ---- stage s1 rows [h-1, h, h+1] into LDS (zero-fill border rows) ----
    const int row_lo   = (h == 0) ? 0 : h - 1;
    const int row_hi   = (h == HZ - 1) ? HZ - 1 : h + 1;
    const int nrows    = row_hi - row_lo + 1;
    const int dst_row0 = row_lo - (h - 1);          // 1 iff h==0, else 0

    if (h == 0)
        for (int idx = tid; idx < ROWH / 8; idx += 256)
            ((h8*)tileA)[idx] = (h8){};             // zero tile row 0
    if (h == HZ - 1)
        for (int idx = tid; idx < ROWH / 8; idx += 256)
            ((h8*)tileA)[2 * (ROWH / 8) + idx] = (h8){}; // zero tile row 2

    const _Float16* srcrow = s1 + (size_t)(b * HZ + row_lo) * ROWH;
    _Float16*       dstrow = tileA + dst_row0 * ROWH;

#if defined(HAVE_TDM)
    // Tensor Data Mover: one 2D-tile DMA per block (nrows x 4096 f16 elements)
    if (tid == 0) {
        unsigned long long ga = (unsigned long long)(uintptr_t)srcrow;
        unsigned int lds_base = (unsigned int)(uintptr_t)dstrow;
        u32x4 g0;
        g0.x = 1u;                                   // count=1, user D#
        g0.y = lds_base;                             // lds_addr (bytes)
        g0.z = (unsigned int)ga;                     // global_addr[31:0]
        g0.w = ((unsigned int)(ga >> 32) & 0x01FFFFFFu) | (2u << 30); // type=2
        i32x8 g1;
        g1[0] = 0x00010000;                          // data_size=1 (2B), no pad
        g1[1] = (int)((ROWH & 0xFFFF) << 16);        // tensor_dim0[15:0]
        g1[2] = (int)(nrows << 16);                  // dim0[31:16]=0, dim1 lo
        g1[3] = (int)((ROWH & 0xFFFF) << 16);        // dim1 hi=0, tile_dim0
        g1[4] = nrows;                               // tile_dim1, tile_dim2=0
        g1[5] = ROWH;                                // tensor_dim0_stride lo32
        g1[6] = (int)((ROWH & 0xFFFF) << 16);        // stride0 hi=0, stride1 lo
        g1[7] = 0;                                   // stride1 hi
        i32x4 zz = (i32x4)0;
        i32x8 z8 = (i32x8)0;
        __builtin_amdgcn_tensor_load_to_lds(g0, g1, zz, zz, z8, 0);
    }
    __builtin_amdgcn_s_wait_tensorcnt((unsigned short)0);
#elif defined(HAVE_ASYNC)
    for (int idx = tid; idx < nrows * (ROWH / 8); idx += 256)
        __builtin_amdgcn_global_load_async_to_lds_b128(
            (const GLOBAL_AS void*)(srcrow + idx * 8),
            (SHARED_AS void*)(dstrow + idx * 8), 0, 0);
    __builtin_amdgcn_s_wait_asynccnt(0);
#else
    for (int idx = tid; idx < nrows * (ROWH / 8); idx += 256)
        ((h8*)dstrow)[idx] = ((const h8*)srcrow)[idx];
#endif
    if (tid < C2) cSum[tid] = 0.f;
    __syncthreads();

    // ---- implicit GEMM: 18 K-steps x 4 WMMAs per wave ----
    v8f acc[4];
    #pragma unroll
    for (int i = 0; i < 4; ++i) acc[i] = (v8f){0.f,0.f,0.f,0.f,0.f,0.f,0.f,0.f};

    for (int tap = 0; tap < 9; ++tap) {
        const int dy = tap / 3 - 1, dx = tap % 3 - 1;
        const int ww = w + dx;
        const bool inb = (ww >= 0) && (ww < WZ);     // rows handled by zero-fill

        #pragma unroll
        for (int kc = 0; kc < 2; ++kc) {
            h8 lo = (h8){}, hi = (h8){};
            if (inb) {
                const _Float16* ap = tileA + (dy + 1) * ROWH + ww * C1
                                   + kc * 32 + ((lane < 16) ? 0 : 8);
                lo = *(const h8*)(ap);
                hi = *(const h8*)(ap + 16);
            }
            v16h a;
            #pragma unroll
            for (int q = 0; q < 8; ++q) { a[q] = lo[q]; a[q + 8] = hi[q]; }

            #pragma unroll
            for (int i = 0; i < 4; ++i) {
                const int frag = (tap * 2 + kc) * 8 + n_half * 4 + i;
                v16h bf = *(const v16h*)(wB + (size_t)frag * 512 + lane * 16);
                acc[i] = __builtin_amdgcn_wmma_f32_16x16x32_f16(
                    false, a, false, bf, (short)0, acc[i], false, false);
            }
        }
    }

    // ---- LIF2 + spike accumulation ----
    const int mbase = (lane < 16) ? 0 : 8;
    #pragma unroll
    for (int i = 0; i < 4; ++i) {
        const int ch = n_half * 64 + i * 16 + (lane & 15);
        const float bias = conv_b[ch];
        float ss = 0.f;
        #pragma unroll
        for (int r = 0; r < 8; ++r) {
            const int pix = pix0 + m_tile * 16 + mbase + r;
            const size_t idx = (size_t)pix * C2 + ch;
            float u = 0.9f * u2[idx] + acc[i][r] + bias;
            float s = (u - 1.0f > 0.0f) ? 1.0f : 0.0f;
            u2[idx] = u - s;
            ss += s;
        }
        atomicAdd(&cSum[ch], ss);
    }
    __syncthreads();

    if (tid < C2) atomicAdd(&sumS2[b * C2 + tid], cSum[tid]);
    if (tid == 0) {
        float tot = 0.f;
        for (int c = 0; c < C2; ++c) tot += cSum[c];
        atomicAdd(&accs[2 * TZ + t], tot);
    }
}

// ---------------------------------------------------------------------------
// k3: head GEMM on the spatially pooled spikes.
// ---------------------------------------------------------------------------
__global__ void k3_head(const float* __restrict__ sumS2,
                        const float* __restrict__ head_w,
                        const float* __restrict__ head_b,
                        float* __restrict__ out, int t) {
    int tid = threadIdx.x;
    if (tid >= BZ * 10) return;
    int b = tid / 10, o = tid % 10;
    float acc = head_b[o];
    const float inv_hw = 1.0f / (float)(HZ * WZ);
    for (int c = 0; c < C2; ++c)
        acc += (sumS2[b * C2 + c] * inv_hw) * head_w[c * 10 + o];
    out[320 + t * 320 + tid] = acc;
}

// ---------------------------------------------------------------------------
// kfin: time reductions. d_out: readout[320] | logits[16*320] | sr[2] | en[1]
// ---------------------------------------------------------------------------
__global__ void kfin(float* __restrict__ out, const float* __restrict__ accs) {
    int tid = threadIdx.x;
    if (tid < 320) {
        float s = 0.f;
        for (int t = 0; t < TZ; ++t) s += out[320 + t * 320 + tid];
        out[tid] = s * (1.0f / TZ);
    } else if (tid == 320) {
        float s = 0.f;
        for (int t = 0; t < TZ; ++t) s += accs[TZ + t];
        out[5440] = s / ((float)TZ * (float)NPIX * (float)C1);
    } else if (tid == 321) {
        float s = 0.f;
        for (int t = 0; t < TZ; ++t) s += accs[2 * TZ + t];
        out[5441] = s / ((float)TZ * (float)NPIX * (float)C2);
    } else if (tid == 322) {
        float s = 0.f;
        for (int t = 0; t < TZ; ++t) s += accs[t];
        out[5442] = s / ((float)TZ * (float)NPIX * 3.0f);
    }
}

extern "C" void kernel_launch(void* const* d_in, const int* in_sizes, int n_in,
                              void* d_out, int out_size, void* d_ws, size_t ws_size,
                              hipStream_t stream) {
    const float* x_seq  = (const float*)d_in[0];
    const float* fuse_w = (const float*)d_in[1];
    const float* fuse_b = (const float*)d_in[2];
    const float* conv_w = (const float*)d_in[3];
    const float* conv_b = (const float*)d_in[4];
    const float* head_w = (const float*)d_in[5];
    const float* head_b = (const float*)d_in[6];
    float* out = (float*)d_out;

    char* ws = (char*)d_ws;
    size_t off = 0;
    float*    u1    = (float*)(ws + off);    off += (size_t)NPIX * C1 * 4;
    float*    u2    = (float*)(ws + off);    off += (size_t)NPIX * C2 * 4;
    _Float16* s1    = (_Float16*)(ws + off); off += (size_t)NPIX * C1 * 2;
    _Float16* wB    = (_Float16*)(ws + off); off += 9 * 2 * 8 * 512 * 2;
    off = (off + 255) & ~(size_t)255;
    float*    sumS2 = (float*)(ws + off);    off += (size_t)BZ * C2 * 4;
    float*    accs  = (float*)(ws + off);    off += 3 * TZ * 4;

    (void)hipMemsetAsync(u1, 0, (size_t)NPIX * C1 * 4, stream);
    (void)hipMemsetAsync(u2, 0, (size_t)NPIX * C2 * 4, stream);
    (void)hipMemsetAsync(accs, 0, 3 * TZ * 4, stream);

    reformat_w_kernel<<<(9 * 2 * 8 * 512 + 255) / 256, 256, 0, stream>>>(conv_w, wB);

    for (int t = 0; t < TZ; ++t) {
        (void)hipMemsetAsync(sumS2, 0, (size_t)BZ * C2 * 4, stream);
        k1_fuse_lif1<<<NPIX / 256, 256, 0, stream>>>(x_seq, fuse_w, fuse_b,
                                                     u1, s1, accs, t);
        k2_conv_lif2<<<NPIX / 64, 256, 0, stream>>>(s1, wB, conv_b,
                                                    u2, sumS2, accs, t);
        k3_head<<<1, 320, 0, stream>>>(sumS2, head_w, head_b, out, t);
    }
    kfin<<<1, 352, 0, stream>>>(out, accs);
}
